// R_83966610636971
// MI455X (gfx1250) — matrix operand order repaired and checked by reading, not compiled, not run on hardware
//
#include <hip/hip_runtime.h>
#include <hip/hip_bf16.h>

typedef _Float16 half_t;
typedef __attribute__((ext_vector_type(16))) _Float16 v16h;
typedef __attribute__((ext_vector_type(4)))  _Float16 v4h;
typedef __attribute__((ext_vector_type(8)))  float    v8f;

// ---------------- LDS arena layout (bytes) ----------------
//   XH   @      0 : f16 [128][48]  swizzled B-layout (x | ones | zeros) (12288)
//   NORM @  12288 : f32 [128][4]                                        ( 2048)
//   XC   @  14336 : f16 [128][96] row-major [h1|x] -> [h2|x]            (24576)
//   P2   @  38912 : f16 [5][128x64 swizzled]                            (81920)
//        (conv1 overlays: M1 f16[128][160] @38912, W1c swz @79872)
//   SCR  @ 120832 :
//        conv2 : W2 swz f16 5*[96x64] @SCR (61440), Atile @SCR+61440 (32768)
//        readout: nnw swz [96x128] @SCR (24576), outb [128][128] @SCR+24576
//                 gl @+57344, wgt @+57856, h3 @+58368, h4 @+58880, h5 @+59392,
//                 red @+59648
// total = 215040 bytes (<= 320KB/WGP, one workgroup per WGP)
constexpr int LDS_BYTES = 215040;

__device__ __forceinline__ v8f vzero8() {
  v8f z;
#pragma unroll
  for (int e = 0; e < 8; ++e) z[e] = 0.0f;
  return z;
}

// ---- B-operand swizzled storage: per 32x16 tile, (k,n) lives at
//      tile*512 + lane*16 + (k%16), lane = n%16 + 16*((k%32)>=16).
//      A full fragment is then 32 contiguous bytes per lane.
__device__ __forceinline__ int bswz(int k, int n, int Ktiles) {
  const int kt = k >> 5, nt = n >> 4;
  const int ln = (n & 15) + (((k >> 4) & 1) << 4);
  return ((nt * Ktiles + kt) << 9) + (ln << 4) + (k & 15);
}
__device__ __forceinline__ v16h load_b_swz(const half_t* B, int Ktiles, int kt,
                                           int nt, int lane) {
  return *(const v16h*)(B + ((nt * Ktiles + kt) << 9) + (lane << 4));
}

// A-operand fragment (16x32 f16) from a row-major LDS tile, row stride ldm.
// Layout used here (same k order as bswz): M = lane%16; each lane half covers one contiguous 16-K run.
__device__ __forceinline__ v16h load_a_frag(const half_t* T, int ldm, int lane) {
  const int m  = lane & 15;
  const int hi = (lane >> 4) << 4;
  const half_t* row = T + m * ldm;
  v16h a;
#pragma unroll
  for (int v = 0; v < 8; ++v) {
    const int k0 = 2 * v + hi;
    a[2 * v]     = row[k0];
    a[2 * v + 1] = row[k0 + 1];
  }
  return a;
}

__device__ __forceinline__ v8f wmma16(v16h a, v16h b, v8f c) {
  return __builtin_amdgcn_wmma_f32_16x16x32_f16(false, a, false, b, (short)0, c,
                                                false, false);
}

__global__ __launch_bounds__(256, 1) void gconv_fused(
    const float* __restrict__ Ag, const float* __restrict__ xg,
    const float* __restrict__ c1w, const float* __restrict__ c1lw,
    const float* __restrict__ c1lb, const float* __restrict__ c2w,
    const float* __restrict__ c2lw, const float* __restrict__ c2lb,
    const float* __restrict__ gw, const float* __restrict__ gb,
    const float* __restrict__ nnwg, const float* __restrict__ nnb,
    const float* __restrict__ l3w, const float* __restrict__ l3b,
    const float* __restrict__ l4w, const float* __restrict__ l4b,
    const float* __restrict__ l5w, const float* __restrict__ l5b,
    float* __restrict__ outg) {
  extern __shared__ char smem[];
  half_t* xh    = (half_t*)(smem);                 // swz B [128 x 48]
  float*  normp = (float*)(smem + 12288);          // [128][4]
  half_t* xc    = (half_t*)(smem + 14336);         // row-major [128][96]
  half_t* p2    = (half_t*)(smem + 38912);         // swz 5 * [128 x 64]
  half_t* m1    = (half_t*)(smem + 38912);         // row-major [128][160]
  half_t* w1c   = (half_t*)(smem + 38912 + 40960); // swz [160 x 64]
  char*   scr   = smem + 120832;
  half_t* w2    = (half_t*)(scr);                  // swz 5 * [96 x 64]
  half_t* atile0= (half_t*)(scr + 61440);          // 8 waves * [4][16][32]
  half_t* nnw   = (half_t*)(scr);                  // swz [96 x 128] (readout)
  half_t* outb  = (half_t*)(scr + 24576);          // row-major [128][128]
  float*  gl    = (float*)(scr + 57344);
  float*  wgt   = (float*)(scr + 57856);
  float*  h3    = (float*)(scr + 58368);
  float*  h4    = (float*)(scr + 58880);
  float*  h5    = (float*)(scr + 59392);
  float*  red   = (float*)(scr + 59648);

  const int tid  = threadIdx.x;
  const int s    = blockIdx.x;
  const int wv   = tid >> 5;
  const int lane = tid & 31;
  const int i0   = wv * 16;
  const int rbase = (lane >> 4) << 3;  // C/D: VGPR r -> row rbase+r, col lane&15

  const float* Asamp = Ag + (size_t)s * (128 * 128 * 5);
  const float* xs    = xg + (size_t)s * (128 * 32);

  // -------- Phase 0: stage x (swz, padded with ones col), xc x-part, W1c ----
  for (int idx = tid; idx < 128 * 48; idx += 256) {
    const int i = idx / 48, a = idx - i * 48;
    const float v = (a < 32) ? xs[i * 32 + a] : (a == 32 ? 1.0f : 0.0f);
    xh[bswz(i, a, 4)] = (half_t)v;
  }
  for (int idx = tid; idx < 128 * 32; idx += 256) {
    const int i = idx >> 5, a = idx & 31;
    xc[i * 96 + 64 + a] = (half_t)xs[idx];
  }
  // W1c row r<128: (c=r/32, a=r%32) -> conv1_w[a][b][c]; rows 128..159: lw
  for (int idx = tid; idx < 160 * 64; idx += 256) {
    const int r = idx >> 6, b = idx & 63;
    const float v = (r < 128) ? c1w[(r & 31) * 256 + b * 4 + (r >> 5)]
                              : c1lw[(r - 128) * 64 + b];
    w1c[bswz(r, b, 5)] = (half_t)v;
  }
  __syncthreads();

  // -------- Phase 1: conv1  M_c = Ar_c @ [x|1], scale by norm, contract W ----
  half_t* atile = atile0 + wv * 2048;  // [4][16][32] halves, row-major per ch
  const int rr = lane >> 1;            // staging row within 16-row block
  const int jb = (lane & 1) << 4;      // staging j-offset (0 or 16)
  {
    v8f acc[4][3];
#pragma unroll
    for (int c = 0; c < 4; ++c)
#pragma unroll
      for (int t = 0; t < 3; ++t) acc[c][t] = vzero8();

    for (int jt = 0; jt < 4; ++jt) {
      const int j0 = jt * 32;
      // stage 16 rows x 32 j x 4 ch: branch-free de-interleave of the 5-ch
      // packed A rows; global_load_b128 in, ds_store_b64 out.
      {
        const float* rowp =
            Asamp + ((size_t)(i0 + rr) * 128 + j0) * 5 + (lane & 1) * 80;
#pragma unroll
        for (int q = 0; q < 4; ++q) {  // 4 j-values (20 floats) per chunk
          float vals[20];
#pragma unroll
          for (int u = 0; u < 5; ++u) {
            const float4 v4 = *(const float4*)(rowp + q * 20 + u * 4);
            vals[u * 4 + 0] = v4.x; vals[u * 4 + 1] = v4.y;
            vals[u * 4 + 2] = v4.z; vals[u * 4 + 3] = v4.w;
          }
#pragma unroll
          for (int c = 0; c < 4; ++c) {
            v4h h;
#pragma unroll
            for (int k = 0; k < 4; ++k) h[k] = (half_t)vals[5 * k + c];
            *(v4h*)(atile + c * 512 + rr * 32 + jb + q * 4) = h;
          }
        }
      }
      v16h bf[3];
#pragma unroll
      for (int t = 0; t < 3; ++t) bf[t] = load_b_swz(xh, 4, jt, t, lane);
#pragma unroll
      for (int c = 0; c < 4; ++c) {
        const v16h af = load_a_frag(atile + c * 512, 32, lane);
#pragma unroll
        for (int t = 0; t < 3; ++t) acc[c][t] = wmma16(af, bf[t], acc[c][t]);
      }
    }
    // rowsum lives in N-tile 2, col 0 (ones column) -> norm
#pragma unroll
    for (int c = 0; c < 4; ++c) {
      if ((lane & 15) == 0) {
#pragma unroll
        for (int r = 0; r < 8; ++r) {
          const int row = i0 + rbase + r;
          normp[row * 4 + c] = 1.0f / (acc[c][2][r] + 1e-7f);
        }
      }
    }
    // M1[i][c*32+a] = norm * M ; M1[i][128+a] = x (from xc)
#pragma unroll
    for (int c = 0; c < 4; ++c)
#pragma unroll
      for (int t = 0; t < 2; ++t)
#pragma unroll
        for (int r = 0; r < 8; ++r) {
          const int row = i0 + rbase + r;
          const int col = c * 32 + t * 16 + (lane & 15);
          m1[row * 160 + col] = (half_t)(normp[row * 4 + c] * acc[c][t][r]);
        }
    for (int q = lane; q < 16 * 32; q += 32) {
      const int r = q >> 5, a = q & 31;
      m1[(i0 + r) * 160 + 128 + a] = xc[(i0 + r) * 96 + 64 + a];
    }
    // contract: out1 = M1(16x160) @ W1c(160x64), then tanh -> xc[:, :64]
    v8f o[4];
#pragma unroll
    for (int t = 0; t < 4; ++t) o[t] = vzero8();
    for (int kk = 0; kk < 5; ++kk) {
      const v16h af = load_a_frag(m1 + i0 * 160 + kk * 32, 160, lane);
#pragma unroll
      for (int t = 0; t < 4; ++t)
        o[t] = wmma16(af, load_b_swz(w1c, 5, kk, t, lane), o[t]);
    }
#pragma unroll
    for (int t = 0; t < 4; ++t)
#pragma unroll
      for (int r = 0; r < 8; ++r) {
        const int row = i0 + rbase + r;
        const int col = t * 16 + (lane & 15);
        xc[row * 96 + col] = (half_t)tanhf(o[t][r] + c1lb[col]);
      }
  }
  __syncthreads();

  // -------- Phase 2: stage W2 (4 rel ch + linear ch), build P2_c = Xc@W2_c ---
  for (int idx = tid; idx < 5 * 96 * 64; idx += 256) {
    const int cc = idx / (96 * 64);
    const int rm = idx - cc * (96 * 64);
    const int a = rm >> 6, b = rm & 63;
    const float v = (cc < 4) ? c2w[a * 256 + b * 4 + cc] : c2lw[a * 64 + b];
    w2[cc * 6144 + bswz(a, b, 3)] = (half_t)v;
  }
  __syncthreads();
  {
    v16h afk[3];
#pragma unroll
    for (int kk = 0; kk < 3; ++kk)
      afk[kk] = load_a_frag(xc + i0 * 96 + kk * 32, 96, lane);
#pragma unroll
    for (int cc = 0; cc < 5; ++cc) {
      v8f p[4];
#pragma unroll
      for (int t = 0; t < 4; ++t) p[t] = vzero8();
#pragma unroll
      for (int kk = 0; kk < 3; ++kk)
#pragma unroll
        for (int t = 0; t < 4; ++t)
          p[t] = wmma16(afk[kk], load_b_swz(w2 + cc * 6144, 3, kk, t, lane),
                        p[t]);
#pragma unroll
      for (int t = 0; t < 4; ++t)
#pragma unroll
        for (int r = 0; r < 8; ++r) {
          const int row = i0 + rbase + r;
          const int col = t * 16 + (lane & 15);
          p2[cc * 8192 + bswz(row, col, 4)] = (half_t)p[t][r];
        }
    }
  }
  __syncthreads();

  // -------- Phase 3: conv2  out2 = Sum_c (norm*Ar_c) @ P2_c + P2_lin + lb ----
  {
    float nsc[4];
#pragma unroll
    for (int c = 0; c < 4; ++c) nsc[c] = normp[(i0 + rr) * 4 + c];
    v8f o2[4];
#pragma unroll
    for (int t = 0; t < 4; ++t) o2[t] = vzero8();
    for (int jt = 0; jt < 4; ++jt) {
      const int j0 = jt * 32;
      {
        const float* rowp =
            Asamp + ((size_t)(i0 + rr) * 128 + j0) * 5 + (lane & 1) * 80;
#pragma unroll
        for (int q = 0; q < 4; ++q) {
          float vals[20];
#pragma unroll
          for (int u = 0; u < 5; ++u) {
            const float4 v4 = *(const float4*)(rowp + q * 20 + u * 4);
            vals[u * 4 + 0] = v4.x; vals[u * 4 + 1] = v4.y;
            vals[u * 4 + 2] = v4.z; vals[u * 4 + 3] = v4.w;
          }
#pragma unroll
          for (int c = 0; c < 4; ++c) {
            v4h h;
#pragma unroll
            for (int k = 0; k < 4; ++k)
              h[k] = (half_t)(vals[5 * k + c] * nsc[c]);
            *(v4h*)(atile + c * 512 + rr * 32 + jb + q * 4) = h;
          }
        }
      }
#pragma unroll
      for (int c = 0; c < 4; ++c) {
        const v16h af = load_a_frag(atile + c * 512, 32, lane);
#pragma unroll
        for (int t = 0; t < 4; ++t)
          o2[t] = wmma16(af, load_b_swz(p2 + c * 8192, 4, jt, t, lane), o2[t]);
      }
    }
#pragma unroll
    for (int t = 0; t < 4; ++t)
#pragma unroll
      for (int r = 0; r < 8; ++r) {
        const int row = i0 + rbase + r;
        const int col = t * 16 + (lane & 15);
        const float v = o2[t][r] +
                        (float)p2[4 * 8192 + bswz(row, col, 4)] + c2lb[col];
        xc[row * 96 + col] = (half_t)tanhf(v);  // xc becomes [h2 | x]
      }
  }
  __syncthreads();

  // -------- Phase 4: gated readout --------
  for (int idx = tid; idx < 96 * 128; idx += 256) {
    const int a = idx >> 7, b = idx & 127;
    nnw[bswz(a, b, 3)] = (half_t)nnwg[idx];
  }
  if (tid < 128) {
    float g = gb[0];
#pragma unroll 4
    for (int a = 0; a < 96; ++a) g += (float)xc[tid * 96 + a] * gw[a];
    gl[tid] = g;
  }
  __syncthreads();
  {
    v16h afk[3];
#pragma unroll
    for (int kk = 0; kk < 3; ++kk)
      afk[kk] = load_a_frag(xc + i0 * 96 + kk * 32, 96, lane);
    v8f ob[8];
#pragma unroll
    for (int t = 0; t < 8; ++t) ob[t] = vzero8();
#pragma unroll
    for (int t = 0; t < 8; ++t)
#pragma unroll
      for (int kk = 0; kk < 3; ++kk)
        ob[t] = wmma16(afk[kk], load_b_swz(nnw, 3, kk, t, lane), ob[t]);
#pragma unroll
    for (int t = 0; t < 8; ++t)
#pragma unroll
      for (int r = 0; r < 8; ++r) {
        const int row = i0 + rbase + r;
        const int col = t * 16 + (lane & 15);
        outb[row * 128 + col] = (half_t)tanhf(ob[t][r] + nnb[col]);
      }
  }
  __syncthreads();
  if (tid == 0) {
    float mx = -3.0e38f;
    for (int i = 0; i < 128; ++i) mx = fmaxf(mx, gl[i]);
    red[0] = mx;
  }
  __syncthreads();
  if (tid < 128) wgt[tid] = __expf(gl[tid] - red[0]);
  __syncthreads();
  if (tid == 0) {
    float sm = 0.0f;
    for (int i = 0; i < 128; ++i) sm += wgt[i];
    red[1] = 1.0f / sm;
  }
  __syncthreads();
  if (tid < 128) {
    float pv = 0.0f;
    for (int i = 0; i < 128; ++i) pv += wgt[i] * (float)outb[i * 128 + tid];
    h3[tid] = tanhf(pv * red[1]);
  }
  __syncthreads();
  if (tid < 128) {
    float v = l3b[tid];
    for (int k = 0; k < 128; ++k) v += h3[k] * l3w[k * 128 + tid];
    h4[tid] = tanhf(v);
  }
  __syncthreads();
  if (tid < 64) {
    float v = l4b[tid];
    for (int k = 0; k < 128; ++k) v += h4[k] * l4w[k * 64 + tid];
    h5[tid] = tanhf(v);
  }
  __syncthreads();
  if (tid == 0) {
    float v = l5b[0];
    for (int k = 0; k < 64; ++k) v += h5[k] * l5w[k];
    outg[s] = 1.0f / (1.0f + __expf(-v));
  }
}

extern "C" void kernel_launch(void* const* d_in, const int* in_sizes, int n_in,
                              void* d_out, int out_size, void* d_ws,
                              size_t ws_size, hipStream_t stream) {
  (void)in_sizes; (void)n_in; (void)d_ws; (void)ws_size; (void)out_size;
  const float* A    = (const float*)d_in[0];
  const float* x    = (const float*)d_in[1];
  const float* c1w  = (const float*)d_in[2];
  const float* c1lw = (const float*)d_in[3];
  const float* c1lb = (const float*)d_in[4];
  const float* c2w  = (const float*)d_in[5];
  const float* c2lw = (const float*)d_in[6];
  const float* c2lb = (const float*)d_in[7];
  const float* gw   = (const float*)d_in[8];
  const float* gb   = (const float*)d_in[9];
  const float* nnw  = (const float*)d_in[10];
  const float* nnb  = (const float*)d_in[11];
  const float* l3w  = (const float*)d_in[12];
  const float* l3b  = (const float*)d_in[13];
  const float* l4w  = (const float*)d_in[14];
  const float* l4b  = (const float*)d_in[15];
  const float* l5w  = (const float*)d_in[16];
  const float* l5b  = (const float*)d_in[17];
  float* out = (float*)d_out;
  gconv_fused<<<512, 256, LDS_BYTES, stream>>>(
      A, x, c1w, c1lw, c1lb, c2w, c2lw, c2lb, gw, gb, nnw, nnb, l3w, l3b, l4w,
      l4b, l5w, l5b, out);
}
